// SparseGATLayer_49409303773443
// MI455X (gfx1250) — matrix operand order repaired and checked by reading, not compiled, not run on hardware
//
#include <hip/hip_runtime.h>
#include <hip/hip_bf16.h>

// ---------------------------------------------------------------------------
// Sparse (top-32) multi-head attention layer for MI455X (gfx1250, wave32).
//
// Pipeline (all f16-in / f32-accum WMMA, fused attention in LDS):
//   1. proj_gemm<f32 A>  : Q = x@Wq+bq, K = x@Wk+bk  (f16, row-major [8192][512])
//                          V = x@Wv+bv stored head-transposed Vt[b*8+h][64][1024]
//   2. attn_kernel       : per (b,h,16-row tile): scores in LDS, exact top-32
//                          threshold + softmax, attn@V via WMMA, ds_add_f32
//                          cross-wave reduction, f16 out.
//   3. proj_gemm<f16 A>  : out = attn_out @ Wo + bo (f32)
// ---------------------------------------------------------------------------

typedef __attribute__((ext_vector_type(16))) _Float16 v16h;
typedef __attribute__((ext_vector_type(8)))  float    v8f;

#define D_MODEL 512
#define SEQ     1024
#define BATCH   8
#define HEADS   8
#define DHEAD   64
#define TOPK    32
#define MTOT    (BATCH * SEQ)   // 8192
#define SPITCH  1032            // padded f32 pitch for the 16x1024 score strip

__device__ __forceinline__ v8f wmma_f16(v16h a, v16h b, v8f c) {
  // D(f32 16x16) = A(f16 16x32) * B(f16 32x16) + C
  return __builtin_amdgcn_wmma_f32_16x16x32_f16(
      /*neg_a=*/false, a, /*neg_b=*/false, b,
      /*c_mod=*/(short)0, c, /*reuse_a=*/false, /*reuse_b=*/false);
}

// ---------------------------------------------------------------------------
// Generic projection GEMM: Out[M=8192][N=512] = A[8192][512] @ W[512][512] + b
// Block: 256 threads = 8 waves, block tile 32(M) x 64(N), wave tile 16x16.
// W tile is staged to LDS transposed + converted to f16; A fragments are read
// straight from global (contiguous 16B runs per lane in the wave32 A layout).
// ---------------------------------------------------------------------------
template <bool A_IS_F32, bool OUT_TRANSPOSED_HEADS, bool OUT_IS_F32>
__global__ __launch_bounds__(256) void proj_gemm(
    const void* __restrict__ Aptr,
    const float* __restrict__ W,      // [512][512] row-major f32
    const float* __restrict__ bias,   // [512]
    void* __restrict__ Out) {
  __shared__ _Float16 Wls[64][40];    // [n_local][k_local], +8 halves pad

  const int tid  = threadIdx.x;
  const int wave = tid >> 5;
  const int lane = tid & 31;
  const int g    = lane >> 4;         // half-wave group
  const int ln   = lane & 15;

  const int rows0 = blockIdx.x * 32;
  const int colb0 = blockIdx.y * 64;
  const int trow0 = rows0 + (wave & 1) * 16;
  const int tcol0 = colb0 + (wave >> 1) * 16;

  v8f acc = {0.f, 0.f, 0.f, 0.f, 0.f, 0.f, 0.f, 0.f};

  const int kk_stage = tid >> 3;       // 0..31 (k within tile)
  const int nn_stage = (tid & 7) * 8;  // 0..56 (n within tile)

  for (int kb = 0; kb < D_MODEL; kb += 32) {
    __syncthreads();
    // stage W[kb..kb+32)[colb0..colb0+64) -> Wls[n][k], cvt f32->f16
#pragma unroll
    for (int j = 0; j < 8; ++j) {
      Wls[nn_stage + j][kk_stage] =
          (_Float16)W[(size_t)(kb + kk_stage) * D_MODEL + colb0 + nn_stage + j];
    }
    __syncthreads();

    // A fragment: lane row = trow0+ln, halves: k = kb+g*8+{0..7} and +16
    v16h a;
    if constexpr (A_IS_F32) {
      const float* ap =
          (const float*)Aptr + (size_t)(trow0 + ln) * D_MODEL + kb + g * 8;
      union { v16h v; _Float16 h[16]; } u;
#pragma unroll
      for (int i = 0; i < 8; ++i) u.h[i] = (_Float16)ap[i];
#pragma unroll
      for (int i = 0; i < 8; ++i) u.h[8 + i] = (_Float16)ap[16 + i];
      a = u.v;
    } else {
      const _Float16* ap =
          (const _Float16*)Aptr + (size_t)(trow0 + ln) * D_MODEL + kb + g * 8;
      union { v16h v; uint4 q[2]; } u;
      u.q[0] = *reinterpret_cast<const uint4*>(ap);
      u.q[1] = *reinterpret_cast<const uint4*>(ap + 16);
      a = u.v;
    }

    // B fragment from LDS: lane col = tcol0+ln, halves: k = g*8+{0..7} and +16
    const _Float16* bp = &Wls[(wave >> 1) * 16 + ln][g * 8];
    union { v16h v; uint4 q[2]; } ub;
    ub.q[0] = *reinterpret_cast<const uint4*>(bp);
    ub.q[1] = *reinterpret_cast<const uint4*>(bp + 16);

    acc = wmma_f16(a, ub.v, acc);
  }

  // Epilogue: C layout (VGPR j, lane): row = trow0 + g*8 + j, col = tcol0 + ln
#pragma unroll
  for (int j = 0; j < 8; ++j) {
    const int row = trow0 + g * 8 + j;
    const int col = tcol0 + ln;
    const float v = acc[j] + bias[col];
    if constexpr (OUT_IS_F32) {
      ((float*)Out)[(size_t)row * D_MODEL + col] = v;
    } else if constexpr (OUT_TRANSPOSED_HEADS) {
      // Vt[b*8+h][d][m]
      const int b = row >> 10, m = row & 1023;
      const int h = col >> 6,  d = col & 63;
      ((_Float16*)Out)[(((size_t)(b * HEADS + h) * DHEAD + d) << 10) + m] =
          (_Float16)v;
    } else {
      ((_Float16*)Out)[(size_t)row * D_MODEL + col] = (_Float16)v;
    }
  }
}

// ---------------------------------------------------------------------------
// Fused top-k attention. Block = 256 threads (8 waves) per (b, h, 16-row tile).
//   Phase A: S[16][1024] = (Q_tile @ K^T) / 8  via WMMA (frags direct from HBM)
//   Phase B: exact 32nd-largest threshold per row (32 argmax rounds, shfl_xor)
//            + thresholded softmax, probabilities written back into S.
//   Phase C: O[16][64] = P @ V via WMMA (B-frags contiguous from Vt),
//            cross-wave combine with ds_add_f32, store f16.
// ---------------------------------------------------------------------------
__global__ __launch_bounds__(256) void attn_kernel(
    const _Float16* __restrict__ Q,    // [8192][512]
    const _Float16* __restrict__ Kf,   // [8192][512]
    const _Float16* __restrict__ Vt,   // [B*H][64][1024]
    _Float16* __restrict__ AO) {       // [8192][512]
  extern __shared__ float smem[];
  float* S = smem;                     // [16][SPITCH]
  float* O = smem + 16 * SPITCH;       // [16][64]

  const int tid  = threadIdx.x;
  const int wave = tid >> 5;
  const int lane = tid & 31;
  const int g    = lane >> 4;
  const int ln   = lane & 15;
  const int rt   = blockIdx.x;         // 0..63 row tile
  const int h    = blockIdx.y;
  const int b    = blockIdx.z;
  const int row0 = rt * 16;

  // ---------------- Phase A: scores ----------------
  const _Float16* qrow =
      Q + (size_t)(b * SEQ + row0 + ln) * D_MODEL + h * DHEAD;
  v16h aq0, aq1;
  {
    union { v16h v; uint4 q[2]; } u;
    u.q[0] = *reinterpret_cast<const uint4*>(qrow + g * 8);
    u.q[1] = *reinterpret_cast<const uint4*>(qrow + g * 8 + 16);
    aq0 = u.v;
    u.q[0] = *reinterpret_cast<const uint4*>(qrow + 32 + g * 8);
    u.q[1] = *reinterpret_cast<const uint4*>(qrow + 32 + g * 8 + 16);
    aq1 = u.v;
  }
  for (int ct = wave; ct < 64; ct += 8) {
    const int key = ct * 16 + ln;
    const _Float16* krow =
        Kf + (size_t)(b * SEQ + key) * D_MODEL + h * DHEAD;
    union { v16h v; uint4 q[2]; } ub;
    v8f sc = {0.f, 0.f, 0.f, 0.f, 0.f, 0.f, 0.f, 0.f};
    ub.q[0] = *reinterpret_cast<const uint4*>(krow + g * 8);
    ub.q[1] = *reinterpret_cast<const uint4*>(krow + g * 8 + 16);
    sc = wmma_f16(aq0, ub.v, sc);
    ub.q[0] = *reinterpret_cast<const uint4*>(krow + 32 + g * 8);
    ub.q[1] = *reinterpret_cast<const uint4*>(krow + 32 + g * 8 + 16);
    sc = wmma_f16(aq1, ub.v, sc);
#pragma unroll
    for (int j = 0; j < 8; ++j)
      S[(size_t)(g * 8 + j) * SPITCH + ct * 16 + ln] = sc[j] * 0.125f;
  }
  __syncthreads();

  // ---------------- Phase B: top-32 threshold + softmax ----------------
  for (int r = wave * 2; r < wave * 2 + 2; ++r) {
    float* Srow = S + (size_t)r * SPITCH;
    float vals[32];
#pragma unroll
    for (int i = 0; i < 32; ++i) vals[i] = Srow[lane + i * 32];

    float rowmax = 0.f, thr = 0.f;
    for (int it = 0; it < TOPK; ++it) {
      float lv = -3.0e38f; int ls = 0;
#pragma unroll
      for (int i = 0; i < 32; ++i)
        if (vals[i] > lv) { lv = vals[i]; ls = i; }
      float v = lv;
      int id = (lane << 5) | ls;
#pragma unroll
      for (int off = 16; off > 0; off >>= 1) {
        const float ov = __shfl_xor(v, off);
        const int  oid = __shfl_xor(id, off);
        if (ov > v || (ov == v && oid < id)) { v = ov; id = oid; }
      }
      if (it == 0) rowmax = v;
      thr = v;
      if ((id >> 5) == lane) {   // winner lane retires exactly one element
#pragma unroll
        for (int i = 0; i < 32; ++i)
          if (i == ls) vals[i] = -3.0e38f;
      }
    }

    float sum = 0.f;
#pragma unroll
    for (int i = 0; i < 32; ++i) {
      const float s = Srow[lane + i * 32];
      const float p = (s >= thr) ? __expf(s - rowmax) : 0.0f;
      Srow[lane + i * 32] = p;
      sum += p;
    }
#pragma unroll
    for (int off = 16; off > 0; off >>= 1) sum += __shfl_xor(sum, off);
    const float inv = 1.0f / sum;
#pragma unroll
    for (int i = 0; i < 32; ++i) Srow[lane + i * 32] *= inv;
  }
  __syncthreads();

  // ---------------- Phase C: out = P @ V ----------------
  const int dt = wave & 3;             // d-tile (16 of 64 head dims)
  const int kh = wave >> 2;            // key-range half
  const int d0 = dt * 16;
  const _Float16* vcol =
      Vt + ((size_t)(b * HEADS + h) * DHEAD + d0 + ln) * SEQ;

  v8f oc = {0.f, 0.f, 0.f, 0.f, 0.f, 0.f, 0.f, 0.f};
  for (int s = kh * 16; s < kh * 16 + 16; ++s) {
    const int k0 = s * 32;
    // A fragment from probabilities (f32 LDS -> f16)
    union { v16h v; _Float16 hh[16]; } ua;
    const float* prow = S + (size_t)ln * SPITCH + k0 + g * 8;
#pragma unroll
    for (int i = 0; i < 8; ++i) ua.hh[i] = (_Float16)prow[i];
#pragma unroll
    for (int i = 0; i < 8; ++i) ua.hh[8 + i] = (_Float16)prow[16 + i];
    // B fragment from Vt: keys contiguous along last dim
    union { v16h v; uint4 q[2]; } ub;
    ub.q[0] = *reinterpret_cast<const uint4*>(vcol + k0 + g * 8);
    ub.q[1] = *reinterpret_cast<const uint4*>(vcol + k0 + g * 8 + 16);
    oc = wmma_f16(ua.v, ub.v, oc);
  }
  __syncthreads();

#pragma unroll
  for (int i = tid; i < 16 * 64; i += 256) O[i] = 0.f;
  __syncthreads();
#pragma unroll
  for (int j = 0; j < 8; ++j)
    atomicAdd(&O[(g * 8 + j) * 64 + d0 + ln], oc[j]);   // ds_add_f32
  __syncthreads();

#pragma unroll
  for (int u = 0; u < 4; ++u) {
    const int e = tid * 4 + u;
    const int r = e >> 6, d = e & 63;
    AO[(size_t)(b * SEQ + row0 + r) * D_MODEL + h * DHEAD + d] =
        (_Float16)O[r * 64 + d];
  }
}

// ---------------------------------------------------------------------------
extern "C" void kernel_launch(void* const* d_in, const int* in_sizes, int n_in,
                              void* d_out, int out_size, void* d_ws,
                              size_t ws_size, hipStream_t stream) {
  const float* x  = (const float*)d_in[0];
  const float* Wq = (const float*)d_in[1];
  const float* bq = (const float*)d_in[2];
  const float* Wk = (const float*)d_in[3];
  const float* bk = (const float*)d_in[4];
  const float* Wv = (const float*)d_in[5];
  const float* bv = (const float*)d_in[6];
  const float* Wo = (const float*)d_in[7];
  const float* bo = (const float*)d_in[8];
  float* out = (float*)d_out;

  _Float16* ws = (_Float16*)d_ws;
  const size_t NELEM = (size_t)MTOT * D_MODEL;   // 4,194,304 halves = 8 MB
  _Float16* Qh = ws;
  _Float16* Kh = ws + NELEM;
  _Float16* Vt = ws + 2 * NELEM;
  _Float16* AO = ws + 3 * NELEM;                 // total 32 MB workspace

  const dim3 gg(MTOT / 32, D_MODEL / 64);        // 256 x 8 blocks

  proj_gemm<true, false, false><<<gg, 256, 0, stream>>>(x, Wq, bq, Qh);
  proj_gemm<true, false, false><<<gg, 256, 0, stream>>>(x, Wk, bk, Kh);
  proj_gemm<true, true,  false><<<gg, 256, 0, stream>>>(x, Wv, bv, Vt);

  const size_t SMEM = (size_t)(16 * SPITCH + 16 * 64) * sizeof(float); // ~68.5 KB
  (void)hipFuncSetAttribute(reinterpret_cast<const void*>(attn_kernel),
                            hipFuncAttributeMaxDynamicSharedMemorySize,
                            (int)SMEM);
  attn_kernel<<<dim3(SEQ / 16, HEADS, BATCH), 256, SMEM, stream>>>(Qh, Kh, Vt,
                                                                   AO);

  proj_gemm<false, false, true><<<gg, 256, 0, stream>>>(AO, Wo, bo, out);
}
